// MultiHeadedAttention_65171833749920
// MI455X (gfx1250) — matrix operand order repaired and checked by reading, not compiled, not run on hardware
//
#include <hip/hip_runtime.h>
#include <stdint.h>

// ---------------------------------------------------------------------------
// MultiHeadedAttention forward for MI455X (gfx1250, wave32, WMMA).
// All GEMMs use v_wmma_f32_16x16x32_bf16 (fp32 accum).
// B=4, T=2048, HID=1024, H=16, DH=64.
//
// Pipeline:
//   0) cvt_bf16_kernel: fp32 -> bf16 one-time conversion of X and weights
//   1) qkv_proj_kernel: bf16 GEMM, 32x64 wave tile; Q pre-scaled by
//      (1/sqrt(DH))*log2(e); V stored transposed [B,H,DH,T]
//   2) flash_attn_kernel: K/V tiles staged into LDS with
//      GLOBAL_LOAD_ASYNC_TO_LDS_B128 (double-buffered, s_wait_asynccnt);
//      S^T = K*Q^T (per-lane softmax stats), O = P*V (P A-frag packs from
//      score regs with zero lane exchange)
//   3) out_proj_kernel: bf16 GEMM -> fp32 output + bias
// ---------------------------------------------------------------------------

#define HID 1024
#define NB  4
#define NT  2048
#define NH  16
#define DH  64
#define NTOK (NB * NT)

typedef __attribute__((ext_vector_type(16))) __bf16 v16bf;
typedef __attribute__((ext_vector_type(8)))  float  v8f;

union BFrag { v16bf v; uint32_t u[8]; uint4 q[2]; };
union Acc   { v8f v; float f[8]; };

// round-to-nearest fp32->bf16 pair pack: 2 adds + 1 v_perm_b32
__device__ __forceinline__ uint32_t pack2(float lo, float hi) {
  uint32_t a = __float_as_uint(lo) + 0x8000u;
  uint32_t b = __float_as_uint(hi) + 0x8000u;
  return __builtin_amdgcn_perm(b, a, 0x07060302u);
}
__device__ __forceinline__ uint16_t bf1(float f) {
  return (uint16_t)((__float_as_uint(f) + 0x8000u) >> 16);
}
__device__ __forceinline__ v8f zero8() {
  v8f z = {0.f,0.f,0.f,0.f,0.f,0.f,0.f,0.f}; return z;
}
__device__ __forceinline__ v8f wmma_bf16(v16bf a, v16bf b, v8f c) {
  return __builtin_amdgcn_wmma_f32_16x16x32_bf16(false, a, false, b, (short)0, c,
                                                 false, false);
}

// ---- bf16 fragment loaders (pure b128 loads) -------------------------------
// A-matrix 16x32 bf16: lane holds row M=lane%16; k runs at 8h and 16+8h.
// B-matrix 32x16 bf16: lane holds col N=lane%16; k runs at 8h and 16+8h.
__device__ __forceinline__ v16bf load_a_bf16(const uint16_t* __restrict__ row,
                                             int kk, int half) {
  BFrag r;
  r.q[0] = *(const uint4*)(row + kk + 8 * half);
  r.q[1] = *(const uint4*)(row + kk + 16 + 8 * half);
  return r.v;
}
__device__ __forceinline__ v16bf load_b_bf16(const uint16_t* __restrict__ row,
                                             int kk, int half) {
  BFrag r;
  r.q[0] = *(const uint4*)(row + kk + 8 * half);
  r.q[1] = *(const uint4*)(row + kk + 16 + 8 * half);
  return r.v;
}

// ---------------------------------------------------------------------------
// Kernel 0: one-time fp32 -> bf16 conversion (8 floats / thread, b128 in/out)
// ---------------------------------------------------------------------------
__global__ __launch_bounds__(256) void cvt_bf16_kernel(
    const float* __restrict__ src, uint16_t* __restrict__ dst) {
  const size_t i = ((size_t)blockIdx.x * 256 + threadIdx.x) * 8;
  const float4 f0 = *(const float4*)(src + i);
  const float4 f1 = *(const float4*)(src + i + 4);
  uint4 o;
  o.x = pack2(f0.x, f0.y); o.y = pack2(f0.z, f0.w);
  o.z = pack2(f1.x, f1.y); o.w = pack2(f1.z, f1.w);
  *(uint4*)(dst + i) = o;
}

// ---------------------------------------------------------------------------
// Kernel 1: QKV projections.  X[8192,1024]bf16 @ W^T[1024,1024]bf16 + b.
// blockIdx.y selects {Q,K,V}.  Wave computes a 32x64 tile (2x4 WMMA frags).
// ---------------------------------------------------------------------------
#define QSCALE 0.18033688011112042f  // 0.125 * log2(e)

__global__ __launch_bounds__(256) void qkv_proj_kernel(
    const uint16_t* __restrict__ Xq, const uint16_t* __restrict__ Xk,
    const uint16_t* __restrict__ Xv,
    const uint16_t* __restrict__ Wqb, const uint16_t* __restrict__ Wkb,
    const uint16_t* __restrict__ Wvb,
    const float* __restrict__ bq, const float* __restrict__ bk,
    const float* __restrict__ bv,
    uint16_t* __restrict__ Qb, uint16_t* __restrict__ Kb,
    uint16_t* __restrict__ Vt) {
  const int wave = threadIdx.x >> 5, lane = threadIdx.x & 31;
  const int half = lane >> 4, l16 = lane & 15;
  const int mode = blockIdx.y;                    // 0=Q 1=K 2=V
  const uint16_t* X = (mode == 0) ? Xq : (mode == 1) ? Xk : Xv;
  const uint16_t* W = (mode == 0) ? Wqb : (mode == 1) ? Wkb : Wvb;
  const float* bias = (mode == 0) ? bq : (mode == 1) ? bk : bv;

  const int tile = blockIdx.x * 8 + wave;         // 4096 tiles of 32x64
  const int m0 = (tile >> 4) * 32;
  const int n0 = (tile & 15) * 64;

  const uint16_t* ar0 = X + (size_t)(m0 + l16) * HID;
  const uint16_t* ar1 = X + (size_t)(m0 + 16 + l16) * HID;
  const uint16_t* br[4];
#pragma unroll
  for (int ni = 0; ni < 4; ++ni)
    br[ni] = W + (size_t)(n0 + ni * 16 + l16) * HID;

  Acc acc[2][4];
#pragma unroll
  for (int mi = 0; mi < 2; ++mi)
#pragma unroll
    for (int ni = 0; ni < 4; ++ni) acc[mi][ni].v = zero8();

  for (int kk = 0; kk < HID; kk += 32) {
    v16bf a0 = load_a_bf16(ar0, kk, half);
    v16bf a1 = load_a_bf16(ar1, kk, half);
#pragma unroll
    for (int ni = 0; ni < 4; ++ni) {
      v16bf b = load_b_bf16(br[ni], kk, half);
      acc[0][ni].v = wmma_bf16(a0, b, acc[0][ni].v);
      acc[1][ni].v = wmma_bf16(a1, b, acc[1][ni].v);
    }
  }

  const int bidx = m0 / NT;
  const float sc = (mode == 0) ? QSCALE : 1.0f;
#pragma unroll
  for (int mi = 0; mi < 2; ++mi) {
#pragma unroll
    for (int ni = 0; ni < 4; ++ni) {
      const int n  = n0 + ni * 16 + l16;
      const int hh = n >> 6, dd = n & 63;
      const int tb = (m0 % NT) + mi * 16;
      const float bb = bias[n];
      if (mode < 2) {
        uint16_t* out = (mode == 0) ? Qb : Kb;
        const size_t base = ((size_t)(bidx * NH + hh) * NT) * DH + dd;
#pragma unroll
        for (int v = 0; v < 8; ++v) {
          const int t = tb + v + 8 * half;
          out[base + (size_t)t * DH] = bf1((acc[mi][ni].f[v] + bb) * sc);
        }
      } else {  // V transposed: 8 consecutive t per lane -> one 16B store
        const size_t base =
            ((size_t)(bidx * NH + hh) * DH + dd) * NT + tb + 8 * half;
        uint4 st;
        st.x = pack2(acc[mi][ni].f[0] + bb, acc[mi][ni].f[1] + bb);
        st.y = pack2(acc[mi][ni].f[2] + bb, acc[mi][ni].f[3] + bb);
        st.z = pack2(acc[mi][ni].f[4] + bb, acc[mi][ni].f[5] + bb);
        st.w = pack2(acc[mi][ni].f[6] + bb, acc[mi][ni].f[7] + bb);
        *(uint4*)(Vt + base) = st;
      }
    }
  }
}

// ---------------------------------------------------------------------------
// Async staging of one 32-key step into LDS: K tile 32x64 bf16 (4KB) and
// V^T tile 64x32 bf16 (4KB).  256 threads x 2 async b128 each.
// ---------------------------------------------------------------------------
__device__ __forceinline__ void issue_tile_async(
    const uint16_t* __restrict__ Kh, const uint16_t* __restrict__ Vh, int kt,
    uint16_t* sk, uint16_t* sv, int tid) {
  // K tile: row = key (32), 64 bf16 per row = 8 chunks of 16B
  const int krow = tid >> 3, kcol = (tid & 7) * 8;
  const uint16_t* gk = Kh + (size_t)(kt + krow) * DH + kcol;
  const uint32_t lk = (uint32_t)(uintptr_t)(sk + krow * DH + kcol);
  asm volatile("global_load_async_to_lds_b128 %0, %1, off"
               :: "v"(lk), "v"(gk) : "memory");
  // V^T tile: row = d (64), 32 bf16 per row = 4 chunks of 16B
  const int vrow = tid >> 2, vcol = (tid & 3) * 8;
  const uint16_t* gv = Vh + (size_t)vrow * NT + kt + vcol;
  const uint32_t lv = (uint32_t)(uintptr_t)(sv + vrow * 32 + vcol);
  asm volatile("global_load_async_to_lds_b128 %0, %1, off"
               :: "v"(lv), "v"(gv) : "memory");
}

// ---------------------------------------------------------------------------
// Kernel 2: flash attention.  8 waves/block share (b,h) and the K/V stream;
// tiles double-buffered in LDS via async copies.  One wave per 16-query tile.
// Scores arrive pre-scaled into the log2 domain (QSCALE folded into Q).
// ---------------------------------------------------------------------------
__global__ __launch_bounds__(256) void flash_attn_kernel(
    const uint16_t* __restrict__ Qb, const uint16_t* __restrict__ Kb,
    const uint16_t* __restrict__ Vt, const int* __restrict__ mask,
    uint16_t* __restrict__ Ob) {
  __shared__ uint16_t sK[2][32 * DH];   // [key][d]
  __shared__ uint16_t sV[2][DH * 32];   // [d][key]

  const int tid = threadIdx.x;
  const int wave = tid >> 5, lane = tid & 31;
  const int half = lane >> 4, l16 = lane & 15;
  const int wid = blockIdx.x * 8 + wave;          // 8192 q-tiles
  const int qt = wid & 127;                       // 8 consecutive per block
  const int h  = (wid >> 7) & (NH - 1);           // same (b,h) for the block
  const int b  = wid >> 11;

  const uint16_t* Qh = Qb + ((size_t)(b * NH + h) * NT) * DH;
  const uint16_t* Kh = Kb + ((size_t)(b * NH + h) * NT) * DH;
  const uint16_t* Vh = Vt + ((size_t)(b * NH + h) * DH) * NT;
  const int* mrow = mask + (size_t)b * NT * NT + (size_t)(qt * 16 + l16) * NT;

  // Q as B-operand of S^T (contraction dim = d, split 0..31 / 32..63)
  const uint16_t* qrow = Qh + (size_t)(qt * 16 + l16) * DH;
  const v16bf qf0 = load_b_bf16(qrow, 0, half);
  const v16bf qf1 = load_b_bf16(qrow, 32, half);

  Acc oc[4];
#pragma unroll
  for (int dt = 0; dt < 4; ++dt) oc[dt].v = zero8();
  float M = -1e30f, S = 0.f;

  issue_tile_async(Kh, Vh, 0, sK[0], sV[0], tid);

  const int NITER = NT / 32;
  for (int it = 0; it < NITER; ++it) {
    const int cur = it & 1;
    const int kt = it * 32;
    if (it + 1 < NITER) {
      issue_tile_async(Kh, Vh, kt + 32, sK[cur ^ 1], sV[cur ^ 1], tid);
      asm volatile("s_wait_asynccnt 0x2" ::: "memory");  // tile 'it' landed
    } else {
      asm volatile("s_wait_asynccnt 0x0" ::: "memory");
    }
    __syncthreads();  // all waves' copies of this tile visible

    // --- scores: two 16-key tiles, contraction over DH=64 (2 wmma each)
    const uint16_t* kr0 = sK[cur] + l16 * DH;
    const uint16_t* kr1 = sK[cur] + (16 + l16) * DH;
    v8f sa = wmma_bf16(load_a_bf16(kr0, 0, half), qf0, zero8());
    sa = wmma_bf16(load_a_bf16(kr0, 32, half), qf1, sa);
    v8f sb = wmma_bf16(load_a_bf16(kr1, 0, half), qf0, zero8());
    sb = wmma_bf16(load_a_bf16(kr1, 32, half), qf1, sb);

    // --- mask (lane's 16 vals: keys kt+8h+0..7, kt+16+8h+0..7)
    int4 ma0 = *(const int4*)(mrow + kt + 8 * half);
    int4 ma1 = *(const int4*)(mrow + kt + 8 * half + 4);
    int4 mb0 = *(const int4*)(mrow + kt + 16 + 8 * half);
    int4 mb1 = *(const int4*)(mrow + kt + 16 + 8 * half + 4);
    int mm[16] = {ma0.x, ma0.y, ma0.z, ma0.w, ma1.x, ma1.y, ma1.z, ma1.w,
                  mb0.x, mb0.y, mb0.z, mb0.w, mb1.x, mb1.y, mb1.z, mb1.w};
    Acc A, Bv; A.v = sa; Bv.v = sb;
    float s[16];
#pragma unroll
    for (int i = 0; i < 8; ++i) {
      s[i]     = mm[i]     ? A.f[i]  : -1e30f;
      s[i + 8] = mm[i + 8] ? Bv.f[i] : -1e30f;
    }

    // --- online softmax in log2 domain (per-lane: one query column)
    float lm = s[0];
#pragma unroll
    for (int i = 1; i < 16; ++i) lm = fmaxf(lm, s[i]);
    const float tm = fmaxf(lm, __shfl_xor(lm, 16, 32));
    const float Mn = fmaxf(M, tm);
    const float al = exp2f(M - Mn);
    float p[16], ls = 0.f;
#pragma unroll
    for (int i = 0; i < 16; ++i) { p[i] = exp2f(s[i] - Mn); ls += p[i]; }
    const float ts = ls + __shfl_xor(ls, 16, 32);
    S = S * al + ts;
    M = Mn;

    // --- rescale O rows (row index of acc elem v is v+8*half -> broadcast)
    float ab[8];
#pragma unroll
    for (int v = 0; v < 8; ++v) ab[v] = __shfl(al, v + 8 * half, 32);
#pragma unroll
    for (int dt = 0; dt < 4; ++dt)
#pragma unroll
      for (int v = 0; v < 8; ++v) oc[dt].f[v] *= ab[v];

    // --- P A-fragment: no cross-lane movement needed (layouts line up)
    BFrag pf;
#pragma unroll
    for (int v = 0; v < 4; ++v) {
      pf.u[v]     = pack2(p[2 * v],     p[2 * v + 1]);
      pf.u[v + 4] = pack2(p[8 + 2 * v], p[9 + 2 * v]);
    }
    // --- O += P(16x32) x V(32x64) from LDS V^T tile
#pragma unroll
    for (int dt = 0; dt < 4; ++dt) {
      const uint16_t* vrow = sV[cur] + (dt * 16 + l16) * 32;
      oc[dt].v = wmma_bf16(pf.v, load_b_bf16(vrow, 0, half), oc[dt].v);
    }
    __syncthreads();  // done reading this buffer; next-next issue may overwrite
  }

  // --- normalize and store O as [B*T, HID] bf16
  float rs[8];
#pragma unroll
  for (int v = 0; v < 8; ++v) rs[v] = 1.f / __shfl(S, v + 8 * half, 32);
#pragma unroll
  for (int dt = 0; dt < 4; ++dt) {
#pragma unroll
    for (int v = 0; v < 8; ++v) {
      const int tok = qt * 16 + v + 8 * half;
      Ob[(size_t)(b * NT + tok) * HID + h * DH + dt * 16 + l16] =
          bf1(oc[dt].f[v] * rs[v]);
    }
  }
}

// ---------------------------------------------------------------------------
// Kernel 3: output projection.  O[8192,1024]bf16 @ Wo^T bf16 + bo -> f32.
// ---------------------------------------------------------------------------
__global__ __launch_bounds__(256) void out_proj_kernel(
    const uint16_t* __restrict__ Ob, const uint16_t* __restrict__ Wob,
    const float* __restrict__ bo, float* __restrict__ out) {
  const int wave = threadIdx.x >> 5, lane = threadIdx.x & 31;
  const int half = lane >> 4, l16 = lane & 15;
  const int tile = blockIdx.x * 8 + wave;         // 4096 tiles of 32x64
  const int m0 = (tile >> 4) * 32;
  const int n0 = (tile & 15) * 64;

  const uint16_t* ar0 = Ob + (size_t)(m0 + l16) * HID;
  const uint16_t* ar1 = Ob + (size_t)(m0 + 16 + l16) * HID;
  const uint16_t* br[4];
#pragma unroll
  for (int ni = 0; ni < 4; ++ni)
    br[ni] = Wob + (size_t)(n0 + ni * 16 + l16) * HID;

  Acc acc[2][4];
#pragma unroll
  for (int mi = 0; mi < 2; ++mi)
#pragma unroll
    for (int ni = 0; ni < 4; ++ni) acc[mi][ni].v = zero8();

  for (int kk = 0; kk < HID; kk += 32) {
    v16bf a0 = load_a_bf16(ar0, kk, half);
    v16bf a1 = load_a_bf16(ar1, kk, half);
#pragma unroll
    for (int ni = 0; ni < 4; ++ni) {
      v16bf b = load_b_bf16(br[ni], kk, half);
      acc[0][ni].v = wmma_bf16(a0, b, acc[0][ni].v);
      acc[1][ni].v = wmma_bf16(a1, b, acc[1][ni].v);
    }
  }

#pragma unroll
  for (int mi = 0; mi < 2; ++mi) {
#pragma unroll
    for (int ni = 0; ni < 4; ++ni) {
      const int n = n0 + ni * 16 + l16;
      const float bb = bo[n];
#pragma unroll
      for (int v = 0; v < 8; ++v) {
        const int row = m0 + mi * 16 + v + 8 * half;
        out[(size_t)row * HID + n] = acc[mi][ni].f[v] + bb;
      }
    }
  }
}

// ---------------------------------------------------------------------------
extern "C" void kernel_launch(void* const* d_in, const int* in_sizes, int n_in,
                              void* d_out, int out_size, void* d_ws,
                              size_t ws_size, hipStream_t stream) {
  const float* query = (const float*)d_in[0];
  const float* key_t = (const float*)d_in[1];
  const float* value = (const float*)d_in[2];
  const int*   mask  = (const int*)d_in[3];
  const float* Wq = (const float*)d_in[4];
  const float* bq = (const float*)d_in[5];
  const float* Wk = (const float*)d_in[6];
  const float* bk = (const float*)d_in[7];
  const float* Wv = (const float*)d_in[8];
  const float* bv = (const float*)d_in[9];
  const float* Wo = (const float*)d_in[10];
  const float* bo = (const float*)d_in[11];

  // workspace layout (bf16 buffers), 120 MiB total
  uint16_t* Qb  = (uint16_t*)d_ws;
  uint16_t* Kb  = Qb  + (size_t)NTOK * HID;
  uint16_t* Vt  = Kb  + (size_t)NTOK * HID;
  uint16_t* Ob  = Vt  + (size_t)NTOK * HID;
  uint16_t* Xqb = Ob  + (size_t)NTOK * HID;
  uint16_t* Xkb = Xqb + (size_t)NTOK * HID;
  uint16_t* Xvb = Xkb + (size_t)NTOK * HID;
  uint16_t* Wqb = Xvb + (size_t)NTOK * HID;
  uint16_t* Wkb = Wqb + (size_t)HID * HID;
  uint16_t* Wvb = Wkb + (size_t)HID * HID;
  uint16_t* Wob = Wvb + (size_t)HID * HID;

  const int gX = (NTOK * HID) / 2048;   // 4096 blocks
  const int gW = (HID * HID) / 2048;    // 512 blocks
  cvt_bf16_kernel<<<gX, 256, 0, stream>>>(query, Xqb);
  cvt_bf16_kernel<<<gX, 256, 0, stream>>>(key_t, Xkb);
  cvt_bf16_kernel<<<gX, 256, 0, stream>>>(value, Xvb);
  cvt_bf16_kernel<<<gW, 256, 0, stream>>>(Wq, Wqb);
  cvt_bf16_kernel<<<gW, 256, 0, stream>>>(Wk, Wkb);
  cvt_bf16_kernel<<<gW, 256, 0, stream>>>(Wv, Wvb);
  cvt_bf16_kernel<<<gW, 256, 0, stream>>>(Wo, Wob);

  qkv_proj_kernel<<<dim3(512, 3), 256, 0, stream>>>(
      Xqb, Xkb, Xvb, Wqb, Wkb, Wvb, bq, bk, bv, Qb, Kb, Vt);
  flash_attn_kernel<<<dim3(1024), 256, 0, stream>>>(Qb, Kb, Vt, mask, Ob);
  out_proj_kernel<<<dim3(512), 256, 0, stream>>>(Ob, Wob, bo, (float*)d_out);
}
